// ScaledDotProductAttention_80942953660779
// MI455X (gfx1250) — matrix operand order repaired and checked by reading, not compiled, not run on hardware
//
#include <hip/hip_runtime.h>
#include <hip/hip_bf16.h>

typedef __attribute__((ext_vector_type(16))) _Float16 v16h;
typedef __attribute__((ext_vector_type(8)))  _Float16 v8h;
typedef __attribute__((ext_vector_type(8)))  float    v8f;
typedef __attribute__((ext_vector_type(4)))  float    v4f;

// Problem constants (fixed by the reference harness)
constexpr int S_LEN = 2048;
constexpr int D_DIM = 64;
constexpr int BH    = 2 * 16;      // B*H
constexpr int QB    = 128;         // queries per block = 8 waves * 16 rows
constexpr int NT    = 32;          // keys per iteration

// Reference divides scores by d_k (not sqrt); fold log2(e) in so native
// v_exp_f32 (exp2) needs no extra multiply. Running max kept in log2 domain.
constexpr float SCALE2 = 1.4426950408889634f / 64.0f;

constexpr int KT_STRIDE = 72;      // halves per K-tile row (64 + pad)
constexpr int VT_STRIDE = 40;      // halves per Vt row (32 + pad)
constexpr int PT_STRIDE = 40;      // halves per P row (32 + pad)
constexpr int KT_HALVES = NT * KT_STRIDE;
constexpr int VT_HALVES = D_DIM * VT_STRIDE;

__device__ __forceinline__ float fast_exp2(float x) { return __builtin_amdgcn_exp2f(x); }

// ---- CDNA5 async LDS copy (ASYNCcnt path, CDNA5 ISA) ----
__device__ __forceinline__ void async_copy_b128(unsigned lds_off, unsigned long long gaddr) {
    asm volatile("global_load_async_to_lds_b128 %0, %1, off"
                 :: "v"(lds_off), "v"(gaddr) : "memory");
}
__device__ __forceinline__ void wait_async0() {
    asm volatile("s_wait_asynccnt 0" ::: "memory");
}
__device__ __forceinline__ unsigned lds_off_of(const void* p) {
    // Flat LDS addresses map to LDS offsets through addr[31:0] (ISA §10.2).
    return (unsigned)(size_t)p;
}

// =====================================================================
// Prologue: K -> f16 (row major), V -> f16 transposed [bh][d][s]
// =====================================================================
__global__ __launch_bounds__(256)
void convert_kv_kernel(const float* __restrict__ k, const float* __restrict__ v,
                       _Float16* __restrict__ k16, _Float16* __restrict__ vt16)
{
    __shared__ __align__(16) _Float16 T[D_DIM * VT_STRIDE];   // transposed V tile

    const int bh = blockIdx.y;
    const int n0 = blockIdx.x * 32;
    const int r  = threadIdx.x >> 3;        // 0..31 (s row in tile)
    const int c  = (threadIdx.x & 7) * 8;   // 0,8,...,56 (d col)

    // K: straight convert, coalesced 16B stores
    {
        const float* src = k + ((size_t)bh * S_LEN + n0 + r) * D_DIM + c;
        v4f a = *(const v4f*)src;
        v4f b = *(const v4f*)(src + 4);
        v8h h;
        #pragma unroll
        for (int i = 0; i < 4; ++i) { h[i] = (_Float16)a[i]; h[4 + i] = (_Float16)b[i]; }
        *(v8h*)(k16 + ((size_t)bh * S_LEN + n0 + r) * D_DIM + c) = h;
    }
    // V: convert + transpose through LDS
    {
        const float* src = v + ((size_t)bh * S_LEN + n0 + r) * D_DIM + c;
        v4f a = *(const v4f*)src;
        v4f b = *(const v4f*)(src + 4);
        #pragma unroll
        for (int i = 0; i < 4; ++i) {
            T[(c + i)     * VT_STRIDE + r] = (_Float16)a[i];
            T[(c + 4 + i) * VT_STRIDE + r] = (_Float16)b[i];
        }
    }
    __syncthreads();
    {
        const int d  = threadIdx.x >> 2;        // 0..63
        const int cc = (threadIdx.x & 3) * 8;   // 0,8,16,24
        v8h h = *(const v8h*)&T[d * VT_STRIDE + cc];
        *(v8h*)(vt16 + ((size_t)bh * D_DIM + d) * S_LEN + n0 + cc) = h;
    }
}

// =====================================================================
// Fast path: f16 K / transposed-V in global, async double-buffered LDS
// =====================================================================
__global__ __launch_bounds__(256)
void fa_fast_kernel(const float* __restrict__ q, const _Float16* __restrict__ k16,
                    const _Float16* __restrict__ vt16, const unsigned char* __restrict__ mask,
                    float* __restrict__ out)
{
    __shared__ __align__(16) _Float16 KtS[2 * KT_HALVES];   // 2 x [32 keys][72]
    __shared__ __align__(16) _Float16 VtS[2 * VT_HALVES];   // 2 x [64 d][40]
    __shared__ __align__(16) _Float16 Pl[8 * 16 * PT_STRIDE]; // per-wave P rows [16 q][40]

    const int bh   = blockIdx.y;
    const int lane = threadIdx.x & 31;
    const int wave = threadIdx.x >> 5;
    const int hi   = lane >> 4;
    const int lq   = lane & 15;
    const int q_row = blockIdx.x * QB + wave * 16 + lq;

    // Q fragments (B-operand of S^T = K*Q^T): lane holds 16 contiguous d of row q
    const float* qrow = q + ((size_t)bh * S_LEN + q_row) * D_DIM;
    v16h qf[2];
    #pragma unroll
    for (int f = 0; f < 2; ++f) {
        const int base = 32 * f + 16 * hi;
        #pragma unroll
        for (int j = 0; j < 16; ++j) qf[f][j] = (_Float16)qrow[base + j];
    }

    float m_run = -INFINITY;
    float l_run = 0.0f;
    v8f o[4];
    #pragma unroll
    for (int c = 0; c < 4; ++c) o[c] = v8f{};

    const _Float16* kbh = k16  + (size_t)bh * S_LEN * D_DIM;
    const _Float16* vbh = vt16 + (size_t)bh * D_DIM * S_LEN;
    const unsigned char* mrow = mask + ((size_t)bh * S_LEN + q_row) * (size_t)S_LEN;

    // async staging map: K tile 32x64 halves -> 256 chunks of 16B; Vt tile 64x32 -> 256 chunks
    const int kr = threadIdx.x >> 3, kc = (threadIdx.x & 7) * 8;   // K row/col (halves)
    const int vd = threadIdx.x >> 2, vc = (threadIdx.x & 3) * 8;   // Vt d/key (halves)

    auto issue_tile = [&](int n0, int buf) {
        async_copy_b128(lds_off_of(&KtS[buf * KT_HALVES + kr * KT_STRIDE + kc]),
                        (unsigned long long)(size_t)(kbh + (size_t)(n0 + kr) * D_DIM + kc));
        async_copy_b128(lds_off_of(&VtS[buf * VT_HALVES + vd * VT_STRIDE + vc]),
                        (unsigned long long)(size_t)(vbh + (size_t)vd * S_LEN + n0 + vc));
    };

    issue_tile(0, 0);
    int buf = 0;

    _Float16* prow = &Pl[(wave * 16 + lq) * PT_STRIDE];

    for (int n0 = 0; n0 < S_LEN; n0 += NT) {
        wait_async0();        // this wave's share of tile `buf` has landed
        __syncthreads();      // all waves' shares landed; previous readers of buf^1 done
        if (n0 + NT < S_LEN) issue_tile(n0 + NT, buf ^ 1);

        const _Float16* Kt = &KtS[buf * KT_HALVES];
        const _Float16* Vt = &VtS[buf * VT_HALVES];

        // ---- S^T = K*Q^T : two 16-key subtiles x two 32-d fragments ----
        v8f s0 = v8f{}, s1 = v8f{};
        #pragma unroll
        for (int f = 0; f < 2; ++f) {
            const int cb = 32 * f + 16 * hi;
            v16h kf0, kf1;
            ((v8h*)&kf0)[0] = *(const v8h*)&Kt[lq * KT_STRIDE + cb];
            ((v8h*)&kf0)[1] = *(const v8h*)&Kt[lq * KT_STRIDE + cb + 8];
            s0 = __builtin_amdgcn_wmma_f32_16x16x32_f16(false, kf0, false, qf[f],
                                                        (short)0, s0, false, false);
            ((v8h*)&kf1)[0] = *(const v8h*)&Kt[(16 + lq) * KT_STRIDE + cb];
            ((v8h*)&kf1)[1] = *(const v8h*)&Kt[(16 + lq) * KT_STRIDE + cb + 8];
            s1 = __builtin_amdgcn_wmma_f32_16x16x32_f16(false, kf1, false, qf[f],
                                                        (short)0, s1, false, false);
        }

        // ---- scale (log2 domain) + mask ----
        const unsigned char* mp = mrow + n0;
        const uint2 mw0 = *(const uint2*)(mp + 8 * hi);
        const uint2 mw1 = *(const uint2*)(mp + 16 + 8 * hi);

        float sc0[8], sc1[8];
        float lmax = -INFINITY;
        #pragma unroll
        for (int r = 0; r < 8; ++r) {
            const unsigned w0 = (r < 4) ? mw0.x : mw0.y;
            const unsigned w1 = (r < 4) ? mw1.x : mw1.y;
            const unsigned b0 = (w0 >> (8 * (r & 3))) & 0xffu;
            const unsigned b1 = (w1 >> (8 * (r & 3))) & 0xffu;
            float a = s0[r] * SCALE2;
            float b = s1[r] * SCALE2;
            a = (b0 != 0u) ? a : -INFINITY;
            b = (b1 != 0u) ? b : -INFINITY;
            sc0[r] = a; sc1[r] = b;
            lmax = fmaxf(lmax, fmaxf(a, b));
        }
        lmax = fmaxf(lmax, __shfl_xor(lmax, 16, 32));

        const float m_new  = fmaxf(m_run, lmax);
        const float m_safe = (m_new == -INFINITY) ? 0.0f : m_new;
        const float alpha  = fast_exp2(m_run - m_safe);

        // ---- exp2, pack, stage P row in per-wave LDS slab ----
        float psum = 0.0f;
        v8h pa, pb;
        #pragma unroll
        for (int r = 0; r < 8; ++r) {
            const float p0 = fast_exp2(sc0[r] - m_safe);
            const float p1 = fast_exp2(sc1[r] - m_safe);
            psum += p0 + p1;
            pa[r] = (_Float16)p0;           // keys 8*hi + r        (subtile 0)
            pb[r] = (_Float16)p1;           // keys 16 + 8*hi + r   (subtile 1)
        }
        *(v8h*)(prow + 8 * hi)      = pa;
        *(v8h*)(prow + 16 + 8 * hi) = pb;
        l_run = l_run * alpha + psum;
        m_run = m_new;

        // B-operand of O^T: lane needs P[q][8*hi .. 8*hi+7] and P[q][16+8*hi .. 16+8*hi+7]
        v16h pf;
        ((v8h*)&pf)[0] = *(const v8h*)(prow + 8 * hi);
        ((v8h*)&pf)[1] = *(const v8h*)(prow + 16 + 8 * hi);

        // ---- rescale + O^T += V^T * P^T ----
        #pragma unroll
        for (int c = 0; c < 4; ++c) {
            #pragma unroll
            for (int r = 0; r < 8; ++r) o[c][r] *= alpha;
            v16h vf;
            ((v8h*)&vf)[0] = *(const v8h*)&Vt[(16 * c + lq) * VT_STRIDE + 8 * hi];
            ((v8h*)&vf)[1] = *(const v8h*)&Vt[(16 * c + lq) * VT_STRIDE + 8 * hi + 16];
            o[c] = __builtin_amdgcn_wmma_f32_16x16x32_f16(false, vf, false, pf,
                                                          (short)0, o[c], false, false);
        }
        buf ^= 1;
    }

    const float l_tot = l_run + __shfl_xor(l_run, 16, 32);
    const float inv   = 1.0f / l_tot;

    float* orow = out + ((size_t)bh * S_LEN + q_row) * D_DIM;
    #pragma unroll
    for (int c = 0; c < 4; ++c) {
        const int d0 = 16 * c + 8 * hi;
        v4f lo, hiv;
        #pragma unroll
        for (int r = 0; r < 4; ++r) { lo[r] = o[c][r] * inv; hiv[r] = o[c][4 + r] * inv; }
        *(v4f*)(orow + d0)     = lo;
        *(v4f*)(orow + d0 + 4) = hiv;
    }
}

// =====================================================================
// Fallback (self-contained, f32 staging) — used if workspace too small
// =====================================================================
__global__ __launch_bounds__(256)
void fa_fwd_kernel(const float* __restrict__ q, const float* __restrict__ k,
                   const float* __restrict__ v, const unsigned char* __restrict__ mask,
                   float* __restrict__ out)
{
    __shared__ __align__(16) _Float16 Kt[NT * KT_STRIDE];
    __shared__ __align__(16) _Float16 Vt[D_DIM * VT_STRIDE];

    const int bh   = blockIdx.y;
    const int lane = threadIdx.x & 31;
    const int wave = threadIdx.x >> 5;
    const int hi   = lane >> 4;
    const int lq   = lane & 15;
    const int q_row = blockIdx.x * QB + wave * 16 + lq;

    const float* qrow = q + ((size_t)bh * S_LEN + q_row) * D_DIM;
    v16h qf[2];
    #pragma unroll
    for (int f = 0; f < 2; ++f) {
        const int base = 32 * f + 16 * hi;
        #pragma unroll
        for (int j = 0; j < 16; ++j) qf[f][j] = (_Float16)qrow[base + j];
    }

    float m_run = -INFINITY;
    float l_run = 0.0f;
    v8f o[4];
    #pragma unroll
    for (int c = 0; c < 4; ++c) o[c] = v8f{};

    const float* kbh = k + (size_t)bh * S_LEN * D_DIM;
    const float* vbh = v + (size_t)bh * S_LEN * D_DIM;
    const unsigned char* mrow = mask + ((size_t)bh * S_LEN + q_row) * (size_t)S_LEN;

    const int ldr = threadIdx.x >> 3;
    const int ldc = (threadIdx.x & 7) * 8;

    for (int n0 = 0; n0 < S_LEN; n0 += NT) {
        __syncthreads();
        {
            const float* src = kbh + (size_t)(n0 + ldr) * D_DIM + ldc;
            v4f a = *(const v4f*)src;
            v4f b = *(const v4f*)(src + 4);
            v8h h;
            #pragma unroll
            for (int i = 0; i < 4; ++i) { h[i] = (_Float16)a[i]; h[4 + i] = (_Float16)b[i]; }
            *(v8h*)&Kt[ldr * KT_STRIDE + ldc] = h;
        }
        {
            const float* src = vbh + (size_t)(n0 + ldr) * D_DIM + ldc;
            v4f a = *(const v4f*)src;
            v4f b = *(const v4f*)(src + 4);
            #pragma unroll
            for (int i = 0; i < 4; ++i) {
                Vt[(ldc + i)     * VT_STRIDE + ldr] = (_Float16)a[i];
                Vt[(ldc + 4 + i) * VT_STRIDE + ldr] = (_Float16)b[i];
            }
        }
        __syncthreads();

        v8f s0 = v8f{}, s1 = v8f{};
        #pragma unroll
        for (int f = 0; f < 2; ++f) {
            const int cb = 32 * f + 16 * hi;
            v16h kf0, kf1;
            ((v8h*)&kf0)[0] = *(const v8h*)&Kt[lq * KT_STRIDE + cb];
            ((v8h*)&kf0)[1] = *(const v8h*)&Kt[lq * KT_STRIDE + cb + 8];
            s0 = __builtin_amdgcn_wmma_f32_16x16x32_f16(false, kf0, false, qf[f],
                                                        (short)0, s0, false, false);
            ((v8h*)&kf1)[0] = *(const v8h*)&Kt[(16 + lq) * KT_STRIDE + cb];
            ((v8h*)&kf1)[1] = *(const v8h*)&Kt[(16 + lq) * KT_STRIDE + cb + 8];
            s1 = __builtin_amdgcn_wmma_f32_16x16x32_f16(false, kf1, false, qf[f],
                                                        (short)0, s1, false, false);
        }

        const unsigned char* mp = mrow + n0;
        const uint2 mw0 = *(const uint2*)(mp + 8 * hi);
        const uint2 mw1 = *(const uint2*)(mp + 16 + 8 * hi);

        float sc0[8], sc1[8];
        float lmax = -INFINITY;
        #pragma unroll
        for (int r = 0; r < 8; ++r) {
            const unsigned w0 = (r < 4) ? mw0.x : mw0.y;
            const unsigned w1 = (r < 4) ? mw1.x : mw1.y;
            const unsigned b0 = (w0 >> (8 * (r & 3))) & 0xffu;
            const unsigned b1 = (w1 >> (8 * (r & 3))) & 0xffu;
            float a = s0[r] * SCALE2;
            float b = s1[r] * SCALE2;
            a = (b0 != 0u) ? a : -INFINITY;
            b = (b1 != 0u) ? b : -INFINITY;
            sc0[r] = a; sc1[r] = b;
            lmax = fmaxf(lmax, fmaxf(a, b));
        }
        lmax = fmaxf(lmax, __shfl_xor(lmax, 16, 32));

        const float m_new  = fmaxf(m_run, lmax);
        const float m_safe = (m_new == -INFINITY) ? 0.0f : m_new;
        const float alpha  = fast_exp2(m_run - m_safe);

        float psum = 0.0f;
        v16h pf;
        #pragma unroll
        for (int r = 0; r < 8; ++r) {
            const float p0 = fast_exp2(sc0[r] - m_safe);
            const float p1 = fast_exp2(sc1[r] - m_safe);
            psum += p0 + p1;
            const float x0 = __shfl_xor(p0, 16, 32);
            const float x1 = __shfl_xor(p1, 16, 32);
            if (hi == 0) { pf[r] = (_Float16)p0; pf[8 + r] = (_Float16)x0; }
            else         { pf[r] = (_Float16)x1; pf[8 + r] = (_Float16)p1; }
        }
        l_run = l_run * alpha + psum;
        m_run = m_new;

        #pragma unroll
        for (int c = 0; c < 4; ++c) {
            #pragma unroll
            for (int r = 0; r < 8; ++r) o[c][r] *= alpha;
            v16h vf;
            ((v8h*)&vf)[0] = *(const v8h*)&Vt[(16 * c + lq) * VT_STRIDE + 16 * hi];
            ((v8h*)&vf)[1] = *(const v8h*)&Vt[(16 * c + lq) * VT_STRIDE + 16 * hi + 8];
            o[c] = __builtin_amdgcn_wmma_f32_16x16x32_f16(false, vf, false, pf,
                                                          (short)0, o[c], false, false);
        }
    }

    const float l_tot = l_run + __shfl_xor(l_run, 16, 32);
    const float inv   = 1.0f / l_tot;

    float* orow = out + ((size_t)bh * S_LEN + q_row) * D_DIM;
    #pragma unroll
    for (int c = 0; c < 4; ++c) {
        const int d0 = 16 * c + 8 * hi;
        v4f lo, hiv;
        #pragma unroll
        for (int r = 0; r < 4; ++r) { lo[r] = o[c][r] * inv; hiv[r] = o[c][4 + r] * inv; }
        *(v4f*)(orow + d0)     = lo;
        *(v4f*)(orow + d0 + 4) = hiv;
    }
}

extern "C" void kernel_launch(void* const* d_in, const int* in_sizes, int n_in,
                              void* d_out, int out_size, void* d_ws, size_t ws_size,
                              hipStream_t stream) {
    const float*         q    = (const float*)d_in[0];
    const float*         k    = (const float*)d_in[1];
    const float*         v    = (const float*)d_in[2];
    const unsigned char* mask = (const unsigned char*)d_in[3];
    float*               out  = (float*)d_out;

    const size_t elems = (size_t)BH * S_LEN * D_DIM;
    const size_t need  = 2 * elems * sizeof(_Float16);   // K16 + Vt16 (~16.8 MB)

    if (ws_size >= need && d_ws != nullptr) {
        _Float16* k16  = (_Float16*)d_ws;
        _Float16* vt16 = k16 + elems;
        convert_kv_kernel<<<dim3(S_LEN / 32, BH), 256, 0, stream>>>(k, v, k16, vt16);
        fa_fast_kernel<<<dim3(S_LEN / QB, BH), 256, 0, stream>>>(q, k16, vt16, mask, out);
    } else {
        fa_fwd_kernel<<<dim3(S_LEN / QB, BH), 256, 0, stream>>>(q, k, v, mask, out);
    }
}
